// GINNet_12025908428954
// MI455X (gfx1250) — matrix operand order repaired and checked by reading, not compiled, not run on hardware
//
#include <hip/hip_runtime.h>
#include <stdint.h>

// ---------------------------------------------------------------------------
// CDNA5 (gfx1250) WMMA types
// ---------------------------------------------------------------------------
typedef __bf16 bf16_t;
typedef bf16_t v16bf __attribute__((ext_vector_type(16)));
typedef float  v8f   __attribute__((ext_vector_type(8)));

union FragBF { v16bf v; uint4 q[2]; };

__device__ __forceinline__ unsigned short f2bf(float f) {
    union { float f; unsigned u; } c; c.f = f;
    unsigned r = c.u + 0x7FFFu + ((c.u >> 16) & 1u);   // round-to-nearest-even
    return (unsigned short)(r >> 16);
}

// ---------------------------------------------------------------------------
// Small utility kernels
// ---------------------------------------------------------------------------
__global__ void k_zero_f32(float* __restrict__ p, long long n) {
    long long i = (long long)blockIdx.x * blockDim.x + threadIdx.x;
    if (i < n) p[i] = 0.0f;
}

__global__ void k_copy_f32(const float* __restrict__ s, float* __restrict__ d, long long n) {
    long long i = (long long)blockIdx.x * blockDim.x + threadIdx.x;
    if (i < n) d[i] = s[i];
}

// GIN aggregation: agg[dst] += x[src], one thread per (edge, feature)
__global__ void k_edge_scatter(const int* __restrict__ src, const int* __restrict__ dst,
                               const float* __restrict__ x, float* __restrict__ agg,
                               long long E, int F) {
    long long i = (long long)blockIdx.x * blockDim.x + threadIdx.x;
    if (i >= E * (long long)F) return;
    long long e = i / F; int f = (int)(i - e * (long long)F);
    int s = src[e], d = dst[e];
    atomicAdd(&agg[(size_t)d * F + f], x[(size_t)s * F + f]);
}

// f32 [rows x F] -> bf16 [rows x Kpad], zero padded
__global__ void k_cast_pad(const float* __restrict__ x, unsigned short* __restrict__ a,
                           int F, int Kpad, long long rows) {
    long long i = (long long)blockIdx.x * blockDim.x + threadIdx.x;
    if (i >= rows * (long long)Kpad) return;
    long long r = i / Kpad; int c = (int)(i - r * (long long)Kpad);
    float v = (c < F) ? x[r * (long long)F + c] : 0.0f;
    a[i] = f2bf(v);
}

// weight (din x dout, row major) -> transposed padded bf16 Wt [Npad x Kpad]
__global__ void k_prep_wt(const float* __restrict__ w, unsigned short* __restrict__ wt,
                          int din, int dout, int Kpad, int Npad) {
    int i = blockIdx.x * blockDim.x + threadIdx.x;
    if (i >= Npad * Kpad) return;
    int n = i / Kpad, k = i - n * Kpad;
    float v = (k < din && n < dout) ? w[(size_t)k * dout + n] : 0.0f;
    wt[i] = f2bf(v);
}

__global__ void k_pool_add(const float* __restrict__ x, const int* __restrict__ batch,
                           float* __restrict__ pool, long long rows, int F) {
    long long i = (long long)blockIdx.x * blockDim.x + threadIdx.x;
    if (i >= rows * (long long)F) return;
    long long n = i / F; int c = (int)(i - n * (long long)F);
    atomicAdd(&pool[(size_t)batch[n] * F + c], x[i]);
}

__global__ void k_count(const int* __restrict__ batch, float* __restrict__ cnt, long long rows) {
    long long i = (long long)blockIdx.x * blockDim.x + threadIdx.x;
    if (i < rows) atomicAdd(&cnt[batch[i]], 1.0f);
}

__global__ void k_pool_fin(const float* __restrict__ pool, const float* __restrict__ cnt,
                           unsigned short* __restrict__ out, int F, int Kpad, int B) {
    int i = blockIdx.x * blockDim.x + threadIdx.x;
    if (i >= B * Kpad) return;
    int b = i / Kpad, c = i - b * Kpad;
    float v = 0.0f;
    if (c < F) v = pool[(size_t)b * F + c] / fmaxf(cnt[b], 1.0f);
    out[i] = f2bf(v);
}

// ---------------------------------------------------------------------------
// WMMA GEMM with 2x2 register blocking:
//   C[M x N] = relu?( A[M x Kpad](bf16) * W[K x N] + bias )
// W supplied transposed+padded as Wt [Npad x Kpad] bf16.
// Each wave owns a 32x32 output macro-tile (4 WMMA accumulators): per k-step
// it loads 2 A-fragments + 2 B-fragments (8x b128) and issues 4
// v_wmma_f32_16x16x32_bf16 — 2 loads per WMMA instead of 4, halving L2 traffic.
// Dual outputs: bf16 (padded, ld = ldB) feeding next GEMM, and/or f32
// (exact, ld = ldF) feeding the next graph layer / final result.
// ---------------------------------------------------------------------------
__global__ void k_wmma_gemm(const unsigned short* __restrict__ A,
                            const unsigned short* __restrict__ Wt,
                            const float* __restrict__ bias, int nExact,
                            unsigned short* __restrict__ outB, int ldB,
                            float* __restrict__ outF, int ldF,
                            int Kpad, int Npad, int relu) {
    const int lane = threadIdx.x & 31;
    const int wave = threadIdx.x >> 5;
    const int colTile = (blockIdx.y * 8 + wave) << 5;   // 32 cols per wave
    if (colTile >= Npad) return;                        // wave-uniform: EXEC stays full
    const int rowTile = blockIdx.x << 5;                // 32 rows per wave

    // A fragment addressing per ISA 16-bit A 16x32 layout:
    //   lane<16  : halves = K[k..k+7]    and K[k+16..k+23]  of row (lane&15)
    //   lane>=16 : halves = K[k+8..k+15] and K[k+24..k+31]
    const int mA   = lane & 15;
    const int kOfA = (lane >> 4) << 3;                  // 0 or 8 halves
    // B fragment: same K layout as A, column (lane&15);
    //   lane<16 -> K[k..k+7] and K[k+16..k+23], lane>=16 -> K[k+8..k+15] and K[k+24..k+31].
    const int nB   = lane & 15;
    const int kOfB = (lane >> 4) << 3;                  // 0 or 8 halves

    const unsigned short* aRow0 = A  + (size_t)(rowTile + mA) * Kpad + kOfA;
    const unsigned short* aRow1 = aRow0 + (size_t)16 * Kpad;
    const unsigned short* bRow0 = Wt + (size_t)(colTile + nB) * Kpad + kOfB;
    const unsigned short* bRow1 = bRow0 + (size_t)16 * Kpad;

    v8f acc00 = {}, acc01 = {}, acc10 = {}, acc11 = {};
    for (int k = 0; k < Kpad; k += 32) {
        FragBF a0, a1, b0, b1;
        a0.q[0] = *(const uint4*)(aRow0 + k);
        a0.q[1] = *(const uint4*)(aRow0 + k + 16);
        a1.q[0] = *(const uint4*)(aRow1 + k);
        a1.q[1] = *(const uint4*)(aRow1 + k + 16);
        b0.q[0] = *(const uint4*)(bRow0 + k);
        b0.q[1] = *(const uint4*)(bRow0 + k + 16);
        b1.q[0] = *(const uint4*)(bRow1 + k);
        b1.q[1] = *(const uint4*)(bRow1 + k + 16);
        __builtin_prefetch(aRow0 + k + 32, 0, 1);       // global_prefetch_b8
        __builtin_prefetch(aRow1 + k + 32, 0, 1);
        acc00 = __builtin_amdgcn_wmma_f32_16x16x32_bf16(
                    false, a0.v, false, b0.v, (short)0, acc00, false, false);
        acc01 = __builtin_amdgcn_wmma_f32_16x16x32_bf16(
                    false, a0.v, false, b1.v, (short)0, acc01, false, false);
        acc10 = __builtin_amdgcn_wmma_f32_16x16x32_bf16(
                    false, a1.v, false, b0.v, (short)0, acc10, false, false);
        acc11 = __builtin_amdgcn_wmma_f32_16x16x32_bf16(
                    false, a1.v, false, b1.v, (short)0, acc11, false, false);
    }

    // C layout: col n = lane&15; VGPR r holds row m = r + 8*(lane>=16)
    auto store = [&](const v8f& acc, int mT, int nT) {
        const int n = nT + nB;
        const float bv = (n < nExact) ? bias[n] : 0.0f;  // padded cols stay exactly 0
        const int mBase = mT + ((lane >> 4) << 3);
#pragma unroll
        for (int r = 0; r < 8; ++r) {
            float v = acc[r] + bv;
            if (relu) v = fmaxf(v, 0.0f);
            const int m = mBase + r;
            if (outB) outB[(size_t)m * ldB + n] = f2bf(v);
            if (outF && n < ldF) outF[(size_t)m * ldF + n] = v;
        }
    };
    store(acc00, rowTile,      colTile);
    store(acc01, rowTile,      colTile + 16);
    store(acc10, rowTile + 16, colTile);
    store(acc11, rowTile + 16, colTile + 16);
}

// ---------------------------------------------------------------------------
// Host orchestration
// ---------------------------------------------------------------------------
extern "C" void kernel_launch(void* const* d_in, const int* in_sizes, int n_in,
                              void* d_out, int out_size, void* d_ws, size_t ws_size,
                              hipStream_t stream) {
    (void)in_sizes; (void)out_size; (void)ws_size;

    const int  N_MOL = 100000; const long long E_MOL = 400000;
    const int  N_PRO = 300000; const long long E_PRO = 1200000;
    const int  B = 512;

    const float* mol_x     = (const float*)d_in[0];
    const int*   mol_ei    = (const int*)  d_in[1];   // [2, E]
    const int*   mol_batch = (const int*)  d_in[2];
    const float* pro_x     = (const float*)d_in[3];
    const int*   pro_ei    = (const int*)  d_in[4];
    const int*   pro_batch = (const int*)  d_in[5];

    // params flattened jax-style (sorted keys at every level), after the 6 arrays:
    // fc1{b,w}, fc2{b,w}, mol{c1[0]{b,w},c1[1]{b,w},c2[0],c2[1],c3[0],c3[1],
    //                         fc_g1{b,w},fc_g2{b,w}}, out{b,w}, pro{...same as mol}
    const float* P[64];
    for (int i = 0; i < n_in && i < 64; ++i) P[i] = (const float*)d_in[i];

    // --- workspace carving -------------------------------------------------
    char* wsb = (char*)d_ws; size_t off = 0;
    auto carve = [&](size_t bytes) -> void* {
        void* p = wsb + off; off += (bytes + 255) & ~(size_t)255; return p;
    };
    const size_t X_EL   = 64800000ull;  // max(100000*328, 300000*216) f32
    const size_t AGG_EL = 32400000ull;  // max(100000*164, 300000*108) f32
    const size_t ABF_EL = 38400000ull;  // max(100000*192, 300000*128) bf16
    const size_t MID_EL = 67200000ull;  // max(100000*352, 300000*224) bf16

    float*          X     = (float*)         carve(X_EL   * 4);
    float*          AGG   = (float*)         carve(AGG_EL * 4);
    unsigned short* ABF   = (unsigned short*)carve(ABF_EL * 2);
    unsigned short* MIDBF = (unsigned short*)carve(MID_EL * 2);
    float*          POOL  = (float*)         carve((size_t)B * 328 * 4);
    float*          CNT   = (float*)         carve((size_t)B * 4);
    unsigned short* PBF   = (unsigned short*)carve((size_t)B * 352 * 2);
    unsigned short* G1BF  = (unsigned short*)carve((size_t)B * 1024 * 2);
    unsigned short* XC    = (unsigned short*)carve((size_t)B * 256 * 2);
    unsigned short* F1BF  = (unsigned short*)carve((size_t)B * 1024 * 2);
    unsigned short* F2BF  = (unsigned short*)carve((size_t)B * 512 * 2);

    auto nb = [](long long n) { return (unsigned)((n + 255) / 256); };

    struct LinW { const float* b; int din, dout, Kpad, Npad; unsigned short* wt; };
    auto mkLin = [&](const float* w_, const float* b_, int din, int dout) {
        LinW L; L.b = b_; L.din = din; L.dout = dout;
        L.Kpad = (din + 31) & ~31; L.Npad = (dout + 31) & ~31;
        L.wt = (unsigned short*)carve((size_t)L.Npad * L.Kpad * 2);
        k_prep_wt<<<nb((long long)L.Npad * L.Kpad), 256, 0, stream>>>(
            w_, L.wt, din, dout, L.Kpad, L.Npad);
        return L;
    };

    LinW fc1  = mkLin(P[7],  P[6],  256, 1024);
    LinW fc2  = mkLin(P[9],  P[8],  1024, 512);
    LinW molL[6] = { mkLin(P[11], P[10],  82,  82), mkLin(P[13], P[12],  82,  82),
                     mkLin(P[15], P[14],  82, 164), mkLin(P[17], P[16], 164, 164),
                     mkLin(P[19], P[18], 164, 328), mkLin(P[21], P[20], 328, 328) };
    LinW molG1 = mkLin(P[23], P[22], 328, 1024);
    LinW molG2 = mkLin(P[25], P[24], 1024, 128);
    LinW lout  = mkLin(P[27], P[26], 512, 1);
    LinW proL[6] = { mkLin(P[29], P[28],  54,  54), mkLin(P[31], P[30],  54,  54),
                     mkLin(P[33], P[32],  54, 108), mkLin(P[35], P[34], 108, 108),
                     mkLin(P[37], P[36], 108, 216), mkLin(P[39], P[38], 216, 216) };
    LinW proG1 = mkLin(P[41], P[40], 216, 1024);
    LinW proG2 = mkLin(P[43], P[42], 1024, 128);

    // M rows are all multiples of 32 (100000, 300000, 512); each wave covers a
    // 32x32 macro-tile, 8 waves/block -> 256 output columns per block.
    auto gemm = [&](const unsigned short* A, const LinW& L, int Mrows,
                    unsigned short* outB, int ldB, float* outF, int ldF, int relu) {
        dim3 grid((unsigned)(Mrows / 32), (unsigned)((L.Npad + 255) / 256));
        k_wmma_gemm<<<grid, 256, 0, stream>>>(A, L.wt, L.b, L.dout,
                                              outB, ldB, outF, ldF,
                                              L.Kpad, L.Npad, relu);
    };

    auto runBranch = [&](const float* x0, const int* ei, const int* batch,
                         int Nn, long long Ee, int f, LinW* L,
                         const LinW& G1w, const LinW& G2w, unsigned short* xcOut) {
        const float* xin = x0;
        int dims[4] = { f, f, 2 * f, 4 * f };
        for (int c = 0; c < 3; ++c) {
            int fin = dims[c];
            const LinW& l1 = L[2 * c];
            const LinW& l2 = L[2 * c + 1];
            long long nel = (long long)Nn * fin;
            k_copy_f32<<<nb(nel), 256, 0, stream>>>(xin, AGG, nel);            // agg = x
            k_edge_scatter<<<nb(Ee * fin), 256, 0, stream>>>(ei, ei + Ee,      // agg += x[src]
                                                             xin, AGG, Ee, fin);
            k_cast_pad<<<nb((long long)Nn * l1.Kpad), 256, 0, stream>>>(
                AGG, ABF, fin, l1.Kpad, Nn);
            gemm(ABF, l1, Nn, MIDBF, l1.Npad, nullptr, 0, 1);                  // relu(h@W1+b1)
            gemm(MIDBF, l2, Nn, nullptr, 0, X, l2.dout, 1);                    // relu(.@W2+b2)
            xin = X;
        }
        int f4 = dims[3];
        k_zero_f32<<<nb((long long)B * f4), 256, 0, stream>>>(POOL, (long long)B * f4);
        k_zero_f32<<<nb(B), 256, 0, stream>>>(CNT, B);
        k_pool_add<<<nb((long long)Nn * f4), 256, 0, stream>>>(X, batch, POOL, Nn, f4);
        k_count<<<nb(Nn), 256, 0, stream>>>(batch, CNT, Nn);
        int Kpg = (f4 + 31) & ~31;
        k_pool_fin<<<nb((long long)B * Kpg), 256, 0, stream>>>(POOL, CNT, PBF, f4, Kpg, B);
        gemm(PBF, G1w, B, G1BF, G1w.Npad, nullptr, 0, 1);                      // fc_g1 + relu
        gemm(G1BF, G2w, B, xcOut, 256, nullptr, 0, 0);                         // fc_g2 -> xc
    };

    runBranch(mol_x, mol_ei, mol_batch, N_MOL, E_MOL, 82, molL, molG1, molG2, XC);
    runBranch(pro_x, pro_ei, pro_batch, N_PRO, E_PRO, 54, proL, proG1, proG2, XC + 128);

    gemm(XC,   fc1, B, F1BF, 1024, nullptr, 0, 1);                             // fc1 + relu
    gemm(F1BF, fc2, B, F2BF,  512, nullptr, 0, 1);                             // fc2 + relu
    gemm(F2BF, lout, B, nullptr, 0, (float*)d_out, 1, 0);                      // out [512,1]
}
